// MultiModalExpertFusion_28656021799527
// MI455X (gfx1250) — matrix operand order repaired and checked by reading, not compiled, not run on hardware
//
#include <hip/hip_runtime.h>

// ---------------------------------------------------------------------------
// Problem constants (MultiModalExpertFusion)
// ---------------------------------------------------------------------------
constexpr int kN   = 8192;   // batch rows
constexpr int kTD  = 768;    // text dim
constexpr int kID  = 1024;   // image dim
constexpr int kD   = 1024;   // shared dim
constexpr int kE   = 8;      // experts
constexpr int kGH  = 512;    // gate hidden
constexpr int k2D  = 2048;   // concat dim

typedef __attribute__((ext_vector_type(16))) __bf16 v16bf;
typedef __attribute__((ext_vector_type(8)))  __bf16 bf16x8;
typedef __attribute__((ext_vector_type(8)))  float  v8f;

__device__ inline void store_val(float* p, float v)  { *p = v; }
__device__ inline void store_val(__bf16* p, float v) { *p = (__bf16)v; }

// Async global->LDS 16B copy (CDNA5, ASYNCcnt-tracked, bypasses VGPRs).
__device__ __forceinline__ void async_copy_b128(unsigned int lds_off,
                                                const void* gptr) {
  unsigned long long ga = (unsigned long long)gptr;
  asm volatile("global_load_async_to_lds_b128 %0, %1, off"
               :: "v"(lds_off), "v"(ga) : "memory");
}
__device__ __forceinline__ void wait_async0() {
  asm volatile("s_wait_asynccnt 0x0" ::: "memory");
}

// ---------------------------------------------------------------------------
// fp32 -> bf16 conversion (activations)
// ---------------------------------------------------------------------------
__global__ void cvt_f32_bf16(const float* __restrict__ src,
                             __bf16* __restrict__ dst, long n) {
  long i = (long)blockIdx.x * blockDim.x + threadIdx.x;
  long stride = (long)gridDim.x * blockDim.x;
  for (; i < n; i += stride) dst[i] = (__bf16)src[i];
}

// ---------------------------------------------------------------------------
// fp32 [K,Nc] -> bf16 [Nc,K] transposed conversion (weights).
// LDS-tiled 32x32 so both global accesses are coalesced.
// Grid: (Nc/32, K/32, batch)
// ---------------------------------------------------------------------------
__global__ __launch_bounds__(256) void cvt_transpose_bf16(
    const float* __restrict__ src, __bf16* __restrict__ dst,
    int K, int Nc, long strideSrc, long strideDst) {
  __shared__ float tile[32][33];
  const int z  = blockIdx.z;
  src += (long)z * strideSrc;
  dst += (long)z * strideDst;
  const int tx = threadIdx.x & 31;
  const int ty = threadIdx.x >> 5;         // 0..7
  const int n0 = blockIdx.x * 32;
  const int k0 = blockIdx.y * 32;
  #pragma unroll
  for (int j = 0; j < 32; j += 8)
    tile[ty + j][tx] = src[(long)(k0 + ty + j) * Nc + n0 + tx];
  __syncthreads();
  #pragma unroll
  for (int j = 0; j < 32; j += 8)
    dst[(long)(n0 + ty + j) * K + k0 + tx] = (__bf16)tile[tx][ty + j];
}

// ---------------------------------------------------------------------------
// Tiled bf16 GEMM with pre-transposed B:
//   C[z] = epilogue(A[z] @ B[z]^T' + bias[z])   where Bt is [Nc,K] row-major
//   A: [M,K] bf16 row-major
//   Bt: [Nc,K] bf16 row-major (weights pre-transposed at conversion time)
//   C: [M,ldc] OutT
// Block tile 128x256, BK=32, 256 threads (8 waves 2x4), wave tile 64x64.
// Double-buffered LDS fed by global_load_async_to_lds_b128.
// Requires M%128==0, Nc%256==0, K%32==0 (true for all call sites here).
// ---------------------------------------------------------------------------
template<typename OutT, bool SILU>
__global__ __launch_bounds__(256) void gemm_bf16(
    const __bf16* __restrict__ A, const __bf16* __restrict__ Bt,
    const float* __restrict__ bias, OutT* __restrict__ C,
    int M, int K, int Nc, int ldc,
    long strideA, long strideB, long strideBias, long strideC)
{
  (void)M; (void)Nc;
  constexpr int BM = 128, BN = 256, BK = 32;
  constexpr int LDA = BK + 8;   // row pitch 80 B (16B aligned)

  __shared__ __bf16 As[2][BM * LDA];   // 20 KB
  __shared__ __bf16 Bs[2][BN * LDA];   // 40 KB  (total 60 KB)

  const int z = blockIdx.z;
  A    += (long)z * strideA;
  Bt   += (long)z * strideB;
  bias += (long)z * strideBias;
  C    += (long)z * strideC;

  const int tid      = threadIdx.x;
  const int wave     = tid >> 5;
  const int lane     = tid & 31;
  const int waveM    = wave >> 2;      // 0..1
  const int waveN    = wave & 3;       // 0..3
  const int halfLane = lane & 15;
  const bool hi      = lane >= 16;
  const int blockRow = blockIdx.y * BM;
  const int blockCol = blockIdx.x * BN;

  v8f acc[4][4];
  #pragma unroll
  for (int mi = 0; mi < 4; ++mi)
    #pragma unroll
    for (int ni = 0; ni < 4; ++ni)
      #pragma unroll
      for (int r = 0; r < 8; ++r) acc[mi][ni][r] = 0.0f;

  // Issue one BK tile of async global->LDS copies into buffer `buf`.
  auto issue_tile = [&](int buf, int kt) {
    // A tile: 128 rows x 32 k = 512 x 16B chunks, 2 per thread
    #pragma unroll
    for (int i = 0; i < 2; ++i) {
      int task  = tid + i * 256;
      int row   = task >> 2;
      int chunk = task & 3;
      const __bf16* g = A + (long)(blockRow + row) * K + kt + chunk * 8;
      unsigned int l = (unsigned int)(uintptr_t)&As[buf][row * LDA + chunk * 8];
      async_copy_b128(l, g);
    }
    // Bt tile: 256 rows x 32 k = 1024 x 16B chunks, 4 per thread
    #pragma unroll
    for (int i = 0; i < 4; ++i) {
      int task  = tid + i * 256;
      int row   = task >> 2;
      int chunk = task & 3;
      const __bf16* g = Bt + (long)(blockCol + row) * K + kt + chunk * 8;
      unsigned int l = (unsigned int)(uintptr_t)&Bs[buf][row * LDA + chunk * 8];
      async_copy_b128(l, g);
    }
  };

  issue_tile(0, 0);
  wait_async0();
  __syncthreads();

  int cur = 0;
  for (int kt = 0; kt < K; kt += BK) {
    if (kt + BK < K) issue_tile(cur ^ 1, kt + BK);

    // A fragments: lane<16 holds K{0..7,16..23}, lane>=16 holds K{8..15,24..31}
    v16bf afrag[4];
    #pragma unroll
    for (int mi = 0; mi < 4; ++mi) {
      const __bf16* p = &As[cur][(waveM * 64 + mi * 16 + halfLane) * LDA
                                 + (hi ? 8 : 0)];
      bf16x8 lo = *(const bf16x8*)p;
      bf16x8 hh = *(const bf16x8*)(p + 16);
      afrag[mi] = __builtin_shufflevector(lo, hh,
          0,1,2,3,4,5,6,7,8,9,10,11,12,13,14,15);
    }
    // B fragments: lane<16 -> col n K{0..7,16..23} ; lane>=16 -> col n K{8..15,24..31}
    v16bf bfrag[4];
    #pragma unroll
    for (int ni = 0; ni < 4; ++ni) {
      const __bf16* p = &Bs[cur][(waveN * 64 + ni * 16 + halfLane) * LDA
                                 + (hi ? 8 : 0)];
      bf16x8 lo = *(const bf16x8*)p;
      bf16x8 hh = *(const bf16x8*)(p + 16);
      bfrag[ni] = __builtin_shufflevector(lo, hh,
          0,1,2,3,4,5,6,7,8,9,10,11,12,13,14,15);
    }
    #pragma unroll
    for (int mi = 0; mi < 4; ++mi)
      #pragma unroll
      for (int ni = 0; ni < 4; ++ni)
        acc[mi][ni] = __builtin_amdgcn_wmma_f32_16x16x32_bf16(
            false, afrag[mi], false, bfrag[ni],
            (short)0, acc[mi][ni], false, false);

    wait_async0();      // next tile landed in LDS
    __syncthreads();    // everyone done reading `cur` / loading `cur^1`
    cur ^= 1;
  }

  // ---- epilogue: bias (+SiLU) and store ----
  #pragma unroll
  for (int mi = 0; mi < 4; ++mi) {
    #pragma unroll
    for (int ni = 0; ni < 4; ++ni) {
      int col  = blockCol + waveN * 64 + ni * 16 + halfLane;
      int row0 = blockRow + waveM * 64 + mi * 16 + (hi ? 8 : 0);
      float bv = bias[col];
      #pragma unroll
      for (int r = 0; r < 8; ++r) {
        float v = acc[mi][ni][r] + bv;
        if (SILU) v = v * (1.0f / (1.0f + __expf(-v)));
        store_val(&C[(long)(row0 + r) * ldc + col], v);
      }
    }
  }
}

// ---------------------------------------------------------------------------
// Gate: logits = gh @ gW2 + gb2 ; w = softmax(logits). One block per row.
// ---------------------------------------------------------------------------
__global__ __launch_bounds__(256) void gate_kernel(
    const __bf16* __restrict__ gh, const float* __restrict__ gW2,
    const float* __restrict__ gb2, float* __restrict__ w)
{
  const int n    = blockIdx.x;
  const int wave = threadIdx.x >> 5;
  const int lane = threadIdx.x & 31;
  const __bf16* row = gh + (long)n * kGH;

  float p = 0.0f;
  for (int k = lane; k < kGH; k += 32)
    p += (float)row[k] * gW2[k * kE + wave];
  #pragma unroll
  for (int off = 16; off > 0; off >>= 1)
    p += __shfl_down(p, off);

  __shared__ float sl[kE];
  if (lane == 0) sl[wave] = p + gb2[wave];
  __syncthreads();

  if (threadIdx.x == 0) {
    float mx = sl[0];
    #pragma unroll
    for (int e = 1; e < kE; ++e) mx = fmaxf(mx, sl[e]);
    float ex[kE], s = 0.0f;
    #pragma unroll
    for (int e = 0; e < kE; ++e) { ex[e] = __expf(sl[e] - mx); s += ex[e]; }
    float inv = 1.0f / s;
    #pragma unroll
    for (int e = 0; e < kE; ++e) w[(long)n * kE + e] = ex[e] * inv;
  }
}

// ---------------------------------------------------------------------------
// Fused per-expert LayerNorm + gate-weighted sum over experts.
// One block (256 threads) per row n; each thread owns 4 columns.
// o layout: [E][N][D] fp32.
// ---------------------------------------------------------------------------
__global__ __launch_bounds__(256) void ln_out_kernel(
    const float* __restrict__ o, const float* __restrict__ w,
    const float* __restrict__ ln_g, const float* __restrict__ ln_b,
    float* __restrict__ out)
{
  const int n    = blockIdx.x;
  const int t    = threadIdx.x;
  const int wave = t >> 5;
  const int lane = t & 31;

  __shared__ float red[2][8];
  float accv[4] = {0.f, 0.f, 0.f, 0.f};

  for (int e = 0; e < kE; ++e) {
    const float* x = o + ((long)e * kN + n) * kD;
    float v[4], s = 0.f, s2 = 0.f;
    #pragma unroll
    for (int j = 0; j < 4; ++j) {
      v[j] = x[t + j * 256];
      s  += v[j];
      s2 += v[j] * v[j];
    }
    #pragma unroll
    for (int off = 16; off > 0; off >>= 1) {
      s  += __shfl_down(s,  off);
      s2 += __shfl_down(s2, off);
    }
    if (lane == 0) { red[0][wave] = s; red[1][wave] = s2; }
    __syncthreads();
    float sum = 0.f, sum2 = 0.f;
    #pragma unroll
    for (int i = 0; i < 8; ++i) { sum += red[0][i]; sum2 += red[1][i]; }
    __syncthreads();

    const float mean = sum * (1.0f / kD);
    const float var  = sum2 * (1.0f / kD) - mean * mean;
    const float rstd = rsqrtf(var + 1e-5f);
    const float we   = w[(long)n * kE + e];
    #pragma unroll
    for (int j = 0; j < 4; ++j) {
      int d = t + j * 256;
      accv[j] += ((v[j] - mean) * rstd * ln_g[e * kD + d] + ln_b[e * kD + d]) * we;
    }
  }
  #pragma unroll
  for (int j = 0; j < 4; ++j)
    out[(long)n * kD + t + j * 256] = accv[j];
}

// ---------------------------------------------------------------------------
// Launch
// ---------------------------------------------------------------------------
extern "C" void kernel_launch(void* const* d_in, const int* in_sizes, int n_in,
                              void* d_out, int out_size, void* d_ws, size_t ws_size,
                              hipStream_t stream) {
  (void)in_sizes; (void)n_in; (void)out_size; (void)ws_size;

  const float* text  = (const float*)d_in[0];   // [N,TD]
  const float* image = (const float*)d_in[1];   // [N,ID]
  const float* Wt    = (const float*)d_in[2];   // [TD,D]
  const float* bt    = (const float*)d_in[3];   // [D]
  const float* Wi    = (const float*)d_in[4];   // [ID,D]
  const float* bi    = (const float*)d_in[5];   // [D]
  const float* eW1   = (const float*)d_in[6];   // [E,2D,D]
  const float* eb1   = (const float*)d_in[7];   // [E,D]
  const float* eW2   = (const float*)d_in[8];   // [E,D,D]
  const float* eb2   = (const float*)d_in[9];   // [E,D]
  const float* ln_g  = (const float*)d_in[10];  // [E,D]
  const float* ln_b  = (const float*)d_in[11];  // [E,D]
  const float* gW1   = (const float*)d_in[12];  // [2D,GH]
  const float* gb1   = (const float*)d_in[13];  // [GH]
  const float* gW2   = (const float*)d_in[14];  // [GH,E]
  const float* gb2   = (const float*)d_in[15];  // [E]

  char* ws = (char*)d_ws;
  size_t off = 0;
  auto take = [&](size_t bytes) -> char* {
    char* p = ws + off;
    off += (bytes + 255) & ~(size_t)255;
    return p;
  };

  __bf16* tfb   = (__bf16*)take((size_t)kN * kTD * 2);
  __bf16* ifb   = (__bf16*)take((size_t)kN * kID * 2);
  __bf16* WtT   = (__bf16*)take((size_t)kTD * kD * 2);   // [D,TD]
  __bf16* WiT   = (__bf16*)take((size_t)kID * kD * 2);   // [D,ID]
  __bf16* gW1T  = (__bf16*)take((size_t)k2D * kGH * 2);  // [GH,2D]
  __bf16* eW1T  = (__bf16*)take((size_t)kE * k2D * kD * 2); // [E][D,2D]
  __bf16* eW2T  = (__bf16*)take((size_t)kE * kD * kD * 2);  // [E][D,D]
  __bf16* cb    = (__bf16*)take((size_t)kN * k2D * 2);
  __bf16* ghb   = (__bf16*)take((size_t)kN * kGH * 2);
  float*  wgate = (float*) take((size_t)kN * kE * 4);
  __bf16* hb    = (__bf16*)take((size_t)kE * kN * kD * 2);
  float*  of    = (float*) take((size_t)kE * kN * kD * 4);

  // --- fp32 -> bf16 conversions (activations straight, weights transposed) ---
  cvt_f32_bf16<<<4096, 256, 0, stream>>>(text,  tfb, (long)kN * kTD);
  cvt_f32_bf16<<<4096, 256, 0, stream>>>(image, ifb, (long)kN * kID);
  cvt_transpose_bf16<<<dim3(kD / 32, kTD / 32, 1), 256, 0, stream>>>(
      Wt, WtT, kTD, kD, 0, 0);
  cvt_transpose_bf16<<<dim3(kD / 32, kID / 32, 1), 256, 0, stream>>>(
      Wi, WiT, kID, kD, 0, 0);
  cvt_transpose_bf16<<<dim3(kGH / 32, k2D / 32, 1), 256, 0, stream>>>(
      gW1, gW1T, k2D, kGH, 0, 0);
  cvt_transpose_bf16<<<dim3(kD / 32, k2D / 32, kE), 256, 0, stream>>>(
      eW1, eW1T, k2D, kD, (long)k2D * kD, (long)k2D * kD);
  cvt_transpose_bf16<<<dim3(kD / 32, kD / 32, kE), 256, 0, stream>>>(
      eW2, eW2T, kD, kD, (long)kD * kD, (long)kD * kD);

  // --- ta = text @ Wt + bt  -> c[:, 0:D] (bf16) ---
  gemm_bf16<__bf16, false><<<dim3(kD / 256, kN / 128, 1), 256, 0, stream>>>(
      tfb, WtT, bt, cb, kN, kTD, kD, k2D, 0, 0, 0, 0);
  // --- ia = image @ Wi + bi -> c[:, D:2D] (bf16) ---
  gemm_bf16<__bf16, false><<<dim3(kD / 256, kN / 128, 1), 256, 0, stream>>>(
      ifb, WiT, bi, cb + kD, kN, kID, kD, k2D, 0, 0, 0, 0);
  // --- gh = silu(c @ gW1 + gb1) (bf16) ---
  gemm_bf16<__bf16, true><<<dim3(kGH / 256, kN / 128, 1), 256, 0, stream>>>(
      cb, gW1T, gb1, ghb, kN, k2D, kGH, kGH, 0, 0, 0, 0);
  // --- w = softmax(gh @ gW2 + gb2) ---
  gate_kernel<<<kN, 256, 0, stream>>>(ghb, gW2, gb2, wgate);
  // --- h[e] = silu(c @ eW1[e] + eb1[e]) (bf16, batched over experts) ---
  gemm_bf16<__bf16, true><<<dim3(kD / 256, kN / 128, kE), 256, 0, stream>>>(
      cb, eW1T, eb1, hb, kN, k2D, kD, kD,
      0, (long)k2D * kD, kD, (long)kN * kD);
  // --- o[e] = h[e] @ eW2[e] + eb2[e] (fp32, batched over experts) ---
  gemm_bf16<float, false><<<dim3(kD / 256, kN / 128, kE), 256, 0, stream>>>(
      hb, eW2T, eb2, of, kN, kD, kD, kD,
      (long)kN * kD, (long)kD * kD, kD, (long)kN * kD);
  // --- per-expert LayerNorm + gate-weighted sum ---
  ln_out_kernel<<<kN, 256, 0, stream>>>(of, wgate, ln_g, ln_b, (float*)d_out);
}
